// MoELayer_85529978733192
// MI455X (gfx1250) — matrix operand order repaired and checked by reading, not compiled, not run on hardware
//
#include <hip/hip_runtime.h>
#include <hip/hip_bf16.h>

#define NTOK 4096   // B*S tokens
#define DIM  1024   // in features
#define NEXP 8
#define HID  4096   // expert hidden
#define TWOH 8192
#define LDT  40     // LDS row stride in halves (80B, 16B-aligned, padded)

typedef __attribute__((ext_vector_type(16))) __bf16 v16bf;
typedef __attribute__((ext_vector_type(8)))  float  v8f;

union FragBF { v16bf v; uint4 q[2]; };

// fp32 -> bf16 with round-half-up (matches RNE except exact ties)
__device__ __forceinline__ unsigned short f2bf(float f) {
    return (unsigned short)((__builtin_bit_cast(unsigned, f) + 0x8000u) >> 16);
}

// pack two floats -> two bf16 in one dword: v_add, v_add, v_perm_b32
__device__ __forceinline__ unsigned pk2bf(float lo, float hi) {
    unsigned a = __builtin_bit_cast(unsigned, lo) + 0x8000u;
    unsigned b = __builtin_bit_cast(unsigned, hi) + 0x8000u;
    // result bytes {b.3, b.2, a.3, a.2}: [15:0]=hi16(a), [31:16]=hi16(b)
    return __builtin_amdgcn_perm(b, a, 0x07060302u);
}

// ---------------------------------------------------------------------------
// Stage 1: gating. One wave (32 lanes) per token.
// ---------------------------------------------------------------------------
__global__ __launch_bounds__(256)
void moe_gate_k(const float* __restrict__ x, const float* __restrict__ Wg,
                int* __restrict__ cnt, float* __restrict__ meanp,
                int* __restrict__ perm, float* __restrict__ pw) {
    int wave = threadIdx.x >> 5;
    int lane = threadIdx.x & 31;
    int n = blockIdx.x * 8 + wave;

    float acc[NEXP];
#pragma unroll
    for (int e = 0; e < NEXP; ++e) acc[e] = 0.f;

    const float* xr = x + (size_t)n * DIM;
    for (int i = lane; i < DIM; i += 32) {
        float xv = xr[i];
        const float* wr = Wg + i * NEXP;
#pragma unroll
        for (int e = 0; e < NEXP; ++e) acc[e] += xv * wr[e];
    }
#pragma unroll
    for (int e = 0; e < NEXP; ++e) {
#pragma unroll
        for (int off = 16; off >= 1; off >>= 1)
            acc[e] += __shfl_xor(acc[e], off, 32);
    }

    if (lane == 0) {
        // full softmax (for aux loss mean_prob)
        float mx = acc[0];
#pragma unroll
        for (int e = 1; e < NEXP; ++e) mx = fmaxf(mx, acc[e]);
        float s = 0.f, p[NEXP];
#pragma unroll
        for (int e = 0; e < NEXP; ++e) { p[e] = __expf(acc[e] - mx); s += p[e]; }
        float inv = 1.f / s;
#pragma unroll
        for (int e = 0; e < NEXP; ++e) atomicAdd(&meanp[e], p[e] * inv);

        // top-2 (desc, index-ordered ties like lax.top_k)
        int i1 = 0;
#pragma unroll
        for (int e = 1; e < NEXP; ++e) if (acc[e] > acc[i1]) i1 = e;
        int i2 = (i1 == 0) ? 1 : 0;
#pragma unroll
        for (int e = 0; e < NEXP; ++e) if (e != i1 && acc[e] > acc[i2]) i2 = e;

        float w1 = 1.f / (1.f + __expf(acc[i2] - acc[i1]));
        float w2 = 1.f - w1;
        int s1 = atomicAdd(&cnt[i1], 1);
        perm[i1 * NTOK + s1] = n;  pw[i1 * NTOK + s1] = w1;
        int s2 = atomicAdd(&cnt[i2], 1);
        perm[i2 * NTOK + s2] = n;  pw[i2 * NTOK + s2] = w2;
    }
}

// ---------------------------------------------------------------------------
// Stage 2: exclusive scan of counts + aux loss.
// ---------------------------------------------------------------------------
__global__ void moe_scan_k(const int* __restrict__ cnt, int* __restrict__ offs,
                           const float* __restrict__ meanp, float* __restrict__ aux_out) {
    if (threadIdx.x == 0) {
        int a = 0;
        for (int e = 0; e < NEXP; ++e) { offs[e] = a; a += cnt[e]; }
        offs[NEXP] = a;
        float aux = 0.f;
        for (int e = 0; e < NEXP; ++e) aux += (float)cnt[e] * meanp[e];
        aux_out[0] = (float)NEXP * aux / ((float)NTOK * (float)NTOK);
    }
}

// ---------------------------------------------------------------------------
// Stage 3: fc1 grouped GEMM + SiLU*val, bf16 WMMA, double-buffered LDS.
// Block = 128 thr (4 waves). Tile: M=64 gathered tokens, N=64 hidden cols
// (computes matching gate AND value columns). K-step 32 over D=1024.
// ---------------------------------------------------------------------------
__global__ __launch_bounds__(128)
void moe_fc1_k(const float* __restrict__ x, const float* __restrict__ w1,
               const float* __restrict__ b1,
               const int* __restrict__ cnt, const int* __restrict__ offs,
               const int* __restrict__ perm, unsigned short* __restrict__ act) {
    int e = blockIdx.z;
    int count = cnt[e];
    int m0 = blockIdx.y * 64;
    if (m0 >= count) return;                 // uniform exit: EXEC stays full
    int n0 = blockIdx.x * 64;                // hidden cols [n0, n0+64)
    int aoff = offs[e];

    __shared__ __align__(16) unsigned short sA[2][64 * LDT];
    __shared__ __align__(16) unsigned short sG[2][64 * LDT];
    __shared__ __align__(16) unsigned short sV[2][64 * LDT];

    int t = threadIdx.x, lane = t & 31, wave = t >> 5;
    int srow = t >> 1, scol = (t & 1) * 16;

    int gidx = m0 + srow; if (gidx >= count) gidx = count - 1;
    const float* xrow = x  + (size_t)perm[e * NTOK + gidx] * DIM + scol;
    const float* grow = w1 + ((size_t)e * TWOH + (size_t)(n0 + srow)) * DIM + scol;
    const float* vrow = w1 + ((size_t)e * TWOH + (size_t)(HID + n0 + srow)) * DIM + scol;

    v8f accg[4], accv[4];
#pragma unroll
    for (int j = 0; j < 4; ++j) {
        int o = n0 + j * 16 + (lane & 15);
        float bg = b1[(size_t)e * TWOH + o];
        float bv = b1[(size_t)e * TWOH + HID + o];
#pragma unroll
        for (int r = 0; r < 8; ++r) { accg[j][r] = bg; accv[j][r] = bv; }
    }

    int afr = wave * 16 + (lane & 15);       // A frag row within tile
    int akb = (lane >> 4) * 16;              // A frag K-half base {0,16}
    int bkb = (lane >> 4) * 16;              // B frag K-half base {0,16}

    float4 rA[4], rG[4], rV[4];
#pragma unroll
    for (int c = 0; c < 4; ++c) {            // prologue: load K-step 0
        rA[c] = *(const float4*)(xrow + 4 * c);
        rG[c] = *(const float4*)(grow + 4 * c);
        rV[c] = *(const float4*)(vrow + 4 * c);
    }

    int buf = 0;
    for (int k0 = 0; k0 < DIM; k0 += 32, buf ^= 1) {
        // commit staged registers -> LDS (packed bf16 pairs via v_perm_b32)
        unsigned* dA = (unsigned*)&sA[buf][srow * LDT + scol];
        unsigned* dG = (unsigned*)&sG[buf][srow * LDT + scol];
        unsigned* dV = (unsigned*)&sV[buf][srow * LDT + scol];
#pragma unroll
        for (int c = 0; c < 4; ++c) {
            dA[2 * c]     = pk2bf(rA[c].x, rA[c].y);
            dA[2 * c + 1] = pk2bf(rA[c].z, rA[c].w);
            dG[2 * c]     = pk2bf(rG[c].x, rG[c].y);
            dG[2 * c + 1] = pk2bf(rG[c].z, rG[c].w);
            dV[2 * c]     = pk2bf(rV[c].x, rV[c].y);
            dV[2 * c + 1] = pk2bf(rV[c].z, rV[c].w);
        }
        // issue next K-step's global loads; latency hides under WMMA
        if (k0 + 32 < DIM) {
#pragma unroll
            for (int c = 0; c < 4; ++c) {
                rA[c] = *(const float4*)(xrow + k0 + 32 + 4 * c);
                rG[c] = *(const float4*)(grow + k0 + 32 + 4 * c);
                rV[c] = *(const float4*)(vrow + k0 + 32 + 4 * c);
            }
        }
        __syncthreads();                     // staged tile visible to all waves

        FragBF a;
        a.q[0] = *(const uint4*)&sA[buf][afr * LDT + akb];
        a.q[1] = *(const uint4*)&sA[buf][afr * LDT + akb + 8];
#pragma unroll
        for (int j = 0; j < 4; ++j) {
            int c = j * 16 + (lane & 15);
            FragBF bg, bv;
            bg.q[0] = *(const uint4*)&sG[buf][c * LDT + bkb];
            bg.q[1] = *(const uint4*)&sG[buf][c * LDT + bkb + 8];
            bv.q[0] = *(const uint4*)&sV[buf][c * LDT + bkb];
            bv.q[1] = *(const uint4*)&sV[buf][c * LDT + bkb + 8];
            accg[j] = __builtin_amdgcn_wmma_f32_16x16x32_bf16(
                false, a.v, false, bg.v, (short)0, accg[j], false, false);
            accv[j] = __builtin_amdgcn_wmma_f32_16x16x32_bf16(
                false, a.v, false, bv.v, (short)0, accv[j], false, false);
        }
        // no second barrier: the collective barrier above guarantees all reads
        // of buffer b completed before the commit that overwrites b (2 iters on)
    }

    // epilogue: act = silu(gate) * val -> bf16
#pragma unroll
    for (int j = 0; j < 4; ++j) {
        int o = n0 + j * 16 + (lane & 15);
#pragma unroll
        for (int r = 0; r < 8; ++r) {
            int i = m0 + wave * 16 + ((lane < 16) ? r : r + 8);
            if (i < count) {
                float g = accg[j][r];
                float v = accv[j][r];
                float a = (g / (1.f + __expf(-g))) * v;
                act[(size_t)(aoff + i) * HID + o] = f2bf(a);
            }
        }
    }
}

// ---------------------------------------------------------------------------
// Stage 4: fc2 grouped GEMM + weighted scatter-add into y.
// A fragments come straight from the bf16 act buffer (fragment-native layout);
// B (w2) tiles double-buffered through LDS.
// ---------------------------------------------------------------------------
__global__ __launch_bounds__(128)
void moe_fc2_k(const unsigned short* __restrict__ act, const float* __restrict__ w2,
               const float* __restrict__ b2,
               const int* __restrict__ cnt, const int* __restrict__ offs,
               const int* __restrict__ perm, const float* __restrict__ pw,
               float* __restrict__ out) {
    int e = blockIdx.z;
    int count = cnt[e];
    int m0 = blockIdx.y * 64;
    if (m0 >= count) return;
    int n0 = blockIdx.x * 64;                // output feature cols
    int aoff = offs[e];

    __shared__ __align__(16) unsigned short sB[2][64 * LDT];

    int t = threadIdx.x, lane = t & 31, wave = t >> 5;
    int srow = t >> 1, scol = (t & 1) * 16;
    const float* w2row = w2 + ((size_t)e * DIM + (size_t)(n0 + srow)) * HID + scol;

    int i = m0 + wave * 16 + (lane & 15);
    int icl = (i < count) ? i : count - 1;
    const unsigned short* arow = act + (size_t)(aoff + icl) * HID;
    int akb = (lane >> 4) * 16;
    int bkb = (lane >> 4) * 16;

    v8f acc[4];
#pragma unroll
    for (int j = 0; j < 4; ++j)
#pragma unroll
        for (int r = 0; r < 8; ++r) acc[j][r] = 0.f;

    float4 rB[4];
#pragma unroll
    for (int c = 0; c < 4; ++c) rB[c] = *(const float4*)(w2row + 4 * c);

    int buf = 0;
    for (int k0 = 0; k0 < HID; k0 += 32, buf ^= 1) {
        unsigned* dB = (unsigned*)&sB[buf][srow * LDT + scol];
#pragma unroll
        for (int c = 0; c < 4; ++c) {
            dB[2 * c]     = pk2bf(rB[c].x, rB[c].y);
            dB[2 * c + 1] = pk2bf(rB[c].z, rB[c].w);
        }
        if (k0 + 32 < HID) {
#pragma unroll
            for (int c = 0; c < 4; ++c)
                rB[c] = *(const float4*)(w2row + k0 + 32 + 4 * c);
        }
        // A fragment straight from global bf16 (fragment-native layout)
        FragBF a;
        a.q[0] = *(const uint4*)&arow[k0 + akb];
        a.q[1] = *(const uint4*)&arow[k0 + akb + 8];
        __syncthreads();
#pragma unroll
        for (int j = 0; j < 4; ++j) {
            int c = j * 16 + (lane & 15);
            FragBF b;
            b.q[0] = *(const uint4*)&sB[buf][c * LDT + bkb];
            b.q[1] = *(const uint4*)&sB[buf][c * LDT + bkb + 8];
            acc[j] = __builtin_amdgcn_wmma_f32_16x16x32_bf16(
                false, a.v, false, b.v, (short)0, acc[j], false, false);
        }
    }

#pragma unroll
    for (int j = 0; j < 4; ++j) {
        int d = n0 + j * 16 + (lane & 15);
        float bias = b2[(size_t)e * DIM + d];
#pragma unroll
        for (int r = 0; r < 8; ++r) {
            int i2 = m0 + wave * 16 + ((lane < 16) ? r : r + 8);
            if (i2 < count) {
                int   tok = perm[e * NTOK + i2];
                float wc  = pw[e * NTOK + i2];
                atomicAdd(&out[(size_t)tok * DIM + d], wc * (acc[j][r] + bias));
            }
        }
    }
}

// ---------------------------------------------------------------------------
// Launch
// ---------------------------------------------------------------------------
extern "C" void kernel_launch(void* const* d_in, const int* in_sizes, int n_in,
                              void* d_out, int out_size, void* d_ws, size_t ws_size,
                              hipStream_t stream) {
    const float* x  = (const float*)d_in[0];
    const float* Wg = (const float*)d_in[1];
    const float* w1 = (const float*)d_in[2];
    const float* b1 = (const float*)d_in[3];
    const float* w2 = (const float*)d_in[4];
    const float* b2 = (const float*)d_in[5];
    float* out = (float*)d_out;

    // workspace layout (bytes):
    //   [0,32)        cnt[8]
    //   [64,100)      offs[9]
    //   [128,160)     meanp[8]
    //   [256, +128K)  perm[8*4096]
    //   [.., +128K)   pw[8*4096]
    //   [262400, ..)  act bf16 [8192 x 4096]  (64 MB)
    char* ws = (char*)d_ws;
    int*   cnt   = (int*)(ws + 0);
    int*   offs  = (int*)(ws + 64);
    float* meanp = (float*)(ws + 128);
    int*   perm  = (int*)(ws + 256);
    float* pw    = (float*)(ws + 256 + sizeof(int) * NEXP * NTOK);
    unsigned short* act = (unsigned short*)(ws + 262400);

    hipMemsetAsync(ws, 0, 256, stream);
    hipMemsetAsync(d_out, 0, sizeof(float) * (size_t)out_size, stream);

    moe_gate_k<<<NTOK / 8, 256, 0, stream>>>(x, Wg, cnt, meanp, perm, pw);
    moe_scan_k<<<1, 32, 0, stream>>>(cnt, offs, meanp, out + (size_t)NTOK * DIM);
    moe_fc1_k<<<dim3(HID / 64, NTOK / 64, NEXP), 128, 0, stream>>>(
        x, w1, b1, cnt, offs, perm, act);
    moe_fc2_k<<<dim3(DIM / 64, NTOK / 64, NEXP), 128, 0, stream>>>(
        act, w2, b2, cnt, offs, perm, pw, out);
}
